// Head_1898375545566
// MI455X (gfx1250) — matrix operand order repaired and checked by reading, not compiled, not run on hardware
//
#include <hip/hip_runtime.h>
#include <stdint.h>

// ---- CDNA5 WMMA types -------------------------------------------------------
typedef __attribute__((ext_vector_type(16))) __bf16        v16bf;
typedef __attribute__((ext_vector_type(8)))  float          v8f;
typedef __attribute__((ext_vector_type(8)))  unsigned short us8;

#define BATCH 8
#define T_SEQ 4096
#define D_EMB 512
#define H_DIM 64

// softmax(x*s) == softmax2(x*s*log2e) with exp2-based accumulation
#define SCALE_LOG2E 0.06377686139013322f  // 512^-0.5 * log2(e)

union BF16Frag {
  v16bf          v;
  __bf16         b[16];
  unsigned short u[16];
  us8            h[2];
};

// ============================================================================
// Kernel 1: QKV projection. X[B*T,512] x W[512,64] -> q,k bf16 [B*T,64],
//           v bf16 transposed [B][64][T]. Q is pre-scaled by 512^-0.5*log2e.
// 128 threads = 4 waves; each wave computes a 16-row tile for q,k,v at once.
// ============================================================================
__global__ __launch_bounds__(128) void qkv_proj_kernel(
    const float* __restrict__ x,  const float* __restrict__ Wq,
    const float* __restrict__ Wk, const float* __restrict__ Wv,
    __bf16* __restrict__ q, __bf16* __restrict__ k,
    __bf16* __restrict__ vT) {
  __shared__ __bf16 sW[3][64][32];  // [mat][n][k-in-chunk]

  const int tid  = threadIdx.x;
  const int wave = tid >> 5;
  const int lane = tid & 31;
  const int nl   = lane & 15;
  const int half = lane >> 4;
  const int g0   = (blockIdx.x * 4 + wave) * 16;  // flat row base in [0, B*T)
  const float* Wm[3] = {Wq, Wk, Wv};

  v8f acc[3][4] = {};

  for (int kc = 0; kc < 16; ++kc) {        // K chunks of 32 over D=512
    __syncthreads();                        // protect previous-iter readers
    // stage bf16 W chunk for all 3 matrices into B-fragment-friendly layout
    for (int idx = tid; idx < 3 * 2048; idx += 128) {
      int mat = idx >> 11;
      int r   = idx & 2047;
      int dl  = r >> 6;        // k within chunk
      int n   = r & 63;        // output column
      sW[mat][n][dl] = (__bf16)Wm[mat][(kc * 32 + dl) * H_DIM + n];
    }
    __syncthreads();

    // A fragment: 16x32 bf16 slice of X (lane m = nl)
    BF16Frag a;
    const float* xr = x + (long)(g0 + nl) * D_EMB + kc * 32;
    #pragma unroll
    for (int e = 0; e < 8; ++e) a.b[e]     = (__bf16)xr[8 * half + e];
    #pragma unroll
    for (int e = 0; e < 8; ++e) a.b[8 + e] = (__bf16)xr[16 + 8 * half + e];

    #pragma unroll
    for (int mat = 0; mat < 3; ++mat) {
      BF16Frag bfr[4];
      #pragma unroll
      for (int t = 0; t < 4; ++t) {
        const __bf16* wrow = &sW[mat][t * 16 + nl][0];
        bfr[t].h[0] = *(const us8*)(wrow + 8 * half);
        bfr[t].h[1] = *(const us8*)(wrow + 16 + 8 * half);
      }
      #pragma unroll
      for (int t = 0; t < 4; ++t)
        acc[mat][t] = __builtin_amdgcn_wmma_f32_16x16x32_bf16(
            false, a.v, false, bfr[t].v, (short)0, acc[mat][t], false, false);
    }
  }

  // store: q,k row-major [B*T,64]; v transposed [B][64][T]
  const int b    = g0 >> 12;           // /4096
  const int trow = g0 & (T_SEQ - 1);
  #pragma unroll
  for (int t = 0; t < 4; ++t) {
    #pragma unroll
    for (int j = 0; j < 8; ++j) {
      const int m = j + 8 * half;
      const int n = t * 16 + nl;
      q[(long)(g0 + m) * H_DIM + n] = (__bf16)(acc[0][t][j] * SCALE_LOG2E);
      k[(long)(g0 + m) * H_DIM + n] = (__bf16)acc[1][t][j];
      vT[((long)b * H_DIM + n) * T_SEQ + (trow + m)] = (__bf16)acc[2][t][j];
    }
  }
}

// ============================================================================
// One 64-key flash-attention tile step. MASK=false for interior tiles (all
// keys strictly below every query row in the 16-row block -> no cndmask work).
// ============================================================================
template <bool MASK>
__device__ __forceinline__ void attn_tile(
    int kb, int q0, int nl, int half,
    const __bf16* __restrict__ kbase, const __bf16* __restrict__ vbase,
    __bf16 (* __restrict__ sPw)[64],                 // [16][64] per-wave P
    const BF16Frag (&aq)[2], v8f (&acc)[4],
    float (&rmax)[8], float (&rsum)[8]) {
  const float NEG_INF = -__builtin_inff();

  // S = Q . K^T for 64 keys (four 16-wide N tiles, two 32-wide h chunks)
  v8f s[4];
  #pragma unroll
  for (int t = 0; t < 4; ++t) {
    BF16Frag bk0, bk1;
    const __bf16* kr = kbase + (long)(kb + t * 16 + nl) * H_DIM;
    bk0.h[0] = *(const us8*)(kr + 16 * half);
    bk0.h[1] = *(const us8*)(kr + 16 * half + 8);
    bk1.h[0] = *(const us8*)(kr + 32 + 16 * half);
    bk1.h[1] = *(const us8*)(kr + 32 + 16 * half + 8);
    v8f z = {};
    z = __builtin_amdgcn_wmma_f32_16x16x32_bf16(false, aq[0].v, false, bk0.v,
                                                (short)0, z, false, false);
    z = __builtin_amdgcn_wmma_f32_16x16x32_bf16(false, aq[1].v, false, bk1.v,
                                                (short)0, z, false, false);
    s[t] = z;
  }

  // online softmax in log2 domain; write P straight to the LDS bounce buffer
  float rescale[8];
  #pragma unroll
  for (int j = 0; j < 8; ++j) {
    float v0 = s[0][j], v1 = s[1][j], v2 = s[2][j], v3 = s[3][j];
    if (MASK) {
      const int qg = q0 + j + 8 * half;
      if (kb + nl      > qg) v0 = NEG_INF;
      if (kb + 16 + nl > qg) v1 = NEG_INF;
      if (kb + 32 + nl > qg) v2 = NEG_INF;
      if (kb + 48 + nl > qg) v3 = NEG_INF;
    }
    float mx = fmaxf(fmaxf(v0, v1), fmaxf(v2, v3));
    #pragma unroll
    for (int msk = 1; msk < 16; msk <<= 1)
      mx = fmaxf(mx, __shfl_xor(mx, msk, 32));
    const float nm = fmaxf(rmax[j], mx);
    const float p0 = exp2f(v0 - nm);
    const float p1 = exp2f(v1 - nm);
    const float p2 = exp2f(v2 - nm);
    const float p3 = exp2f(v3 - nm);
    sPw[j + 8 * half][nl]      = (__bf16)p0;
    sPw[j + 8 * half][16 + nl] = (__bf16)p1;
    sPw[j + 8 * half][32 + nl] = (__bf16)p2;
    sPw[j + 8 * half][48 + nl] = (__bf16)p3;
    float ps = (p0 + p1) + (p2 + p3);
    #pragma unroll
    for (int msk = 1; msk < 16; msk <<= 1) ps += __shfl_xor(ps, msk, 32);
    rescale[j] = exp2f(rmax[j] - nm);
    rmax[j]    = nm;
    rsum[j]    = rsum[j] * rescale[j] + ps;
  }

  #pragma unroll
  for (int ht = 0; ht < 4; ++ht)
    #pragma unroll
    for (int j = 0; j < 8; ++j) acc[ht][j] *= rescale[j];

  // P as A-fragments (two 32-key chunks) from the LDS bounce
  BF16Frag ap[2];
  {
    const __bf16* prow = &sPw[nl][0];
    #pragma unroll
    for (int c = 0; c < 2; ++c) {
      ap[c].h[0] = *(const us8*)(prow + c * 32 + 16 * half);
      ap[c].h[1] = *(const us8*)(prow + c * 32 + 16 * half + 8);
    }
  }

  // acc += P . V  (V transposed: contiguous along key dim)
  #pragma unroll
  for (int ht = 0; ht < 4; ++ht) {
    BF16Frag bv0, bv1;
    const __bf16* vr = vbase + (long)(ht * 16 + nl) * T_SEQ + kb + 16 * half;
    bv0.h[0] = *(const us8*)(vr);
    bv0.h[1] = *(const us8*)(vr + 8);
    bv1.h[0] = *(const us8*)(vr + 32);
    bv1.h[1] = *(const us8*)(vr + 40);
    acc[ht] = __builtin_amdgcn_wmma_f32_16x16x32_bf16(
        false, ap[0].v, false, bv0.v, (short)0, acc[ht], false, false);
    acc[ht] = __builtin_amdgcn_wmma_f32_16x16x32_bf16(
        false, ap[1].v, false, bv1.v, (short)0, acc[ht], false, false);
  }
}

// ============================================================================
// Kernel 2: causal flash attention. One wave per 16-query tile; 64-key steps.
// Interior tiles take the maskless fast path; only the diagonal tile masks.
// ============================================================================
__global__ __launch_bounds__(128) void attn_kernel(
    const __bf16* __restrict__ q, const __bf16* __restrict__ k,
    const __bf16* __restrict__ vT, float* __restrict__ out) {
  __shared__ __bf16 sP[4][16][64];  // per-wave P bounce buffer (2 KB/wave)

  const int tid  = threadIdx.x;
  const int wave = tid >> 5;
  const int lane = tid & 31;
  const int nl   = lane & 15;
  const int half = lane >> 4;
  const int g0   = (blockIdx.x * 4 + wave) * 16;  // flat query row base
  const int b    = g0 >> 12;
  const int q0   = g0 & (T_SEQ - 1);
  const float NEG_INF = -__builtin_inff();

  // Q fragments for the two 32-wide h chunks (Q already carries scale*log2e)
  BF16Frag aq[2];
  {
    const __bf16* qr = q + (long)(g0 + nl) * H_DIM;
    #pragma unroll
    for (int c = 0; c < 2; ++c) {
      aq[c].h[0] = *(const us8*)(qr + c * 32 + 16 * half);
      aq[c].h[1] = *(const us8*)(qr + c * 32 + 16 * half + 8);
    }
  }

  v8f acc[4] = {};
  float rmax[8], rsum[8];
  #pragma unroll
  for (int j = 0; j < 8; ++j) { rmax[j] = NEG_INF; rsum[j] = 0.0f; }

  const __bf16* kbase = k  + (long)b * T_SEQ * H_DIM;
  const __bf16* vbase = vT + (long)b * H_DIM * T_SEQ;
  const int ntile = ((q0 + 15) >> 6) + 1;  // 64-key tiles (causal bound)

  for (int it = 0; it < ntile - 1; ++it) {  // interior: provably unmasked
    const int kb = it * 64;
    __builtin_prefetch(kbase + (long)(kb + 64 + nl) * H_DIM, 0, 1);
    __builtin_prefetch(vbase + (long)nl * T_SEQ + kb + 64, 0, 1);
    attn_tile<false>(kb, q0, nl, half, kbase, vbase, sP[wave], aq, acc, rmax,
                     rsum);
  }
  // diagonal tile: causal mask active
  attn_tile<true>((ntile - 1) * 64, q0, nl, half, kbase, vbase, sP[wave], aq,
                  acc, rmax, rsum);

  // normalize and write fp32 output [B*T, 64]
  float inv[8];
  #pragma unroll
  for (int j = 0; j < 8; ++j) inv[j] = 1.0f / rsum[j];
  #pragma unroll
  for (int ht = 0; ht < 4; ++ht)
    #pragma unroll
    for (int j = 0; j < 8; ++j)
      out[(long)(g0 + j + 8 * half) * H_DIM + ht * 16 + nl] =
          acc[ht][j] * inv[j];
}

// ============================================================================
extern "C" void kernel_launch(void* const* d_in, const int* in_sizes, int n_in,
                              void* d_out, int out_size, void* d_ws,
                              size_t ws_size, hipStream_t stream) {
  const float* x  = (const float*)d_in[0];
  const float* Wq = (const float*)d_in[1];
  const float* Wk = (const float*)d_in[2];
  const float* Wv = (const float*)d_in[3];
  float* out = (float*)d_out;

  const size_t qkv_bytes =
      (size_t)BATCH * T_SEQ * H_DIM * sizeof(__bf16);  // 4 MB each
  __bf16* q  = (__bf16*)d_ws;
  __bf16* k  = (__bf16*)((char*)d_ws + qkv_bytes);
  __bf16* vT = (__bf16*)((char*)d_ws + 2 * qkv_bytes);

  const int blocks = BATCH * T_SEQ / 64;  // 4 waves/block, 16 rows/wave
  qkv_proj_kernel<<<blocks, 128, 0, stream>>>(x, Wq, Wk, Wv, q, k, vT);
  attn_kernel<<<blocks, 128, 0, stream>>>(q, k, vT, out);
}
